// SwinBlock_20693152432505
// MI455X (gfx1250) — matrix operand order repaired and checked by reading, not compiled, not run on hardware
//
#include <hip/hip_runtime.h>

// ---------------- CDNA5 WMMA types ----------------
typedef __attribute__((ext_vector_type(16))) __bf16 v16bf;
typedef __attribute__((ext_vector_type(8)))  float  v8f;

union FragB16 { uint4 q[2]; v16bf v; };   // 32 bytes = one 16-bit A/B WMMA operand

__device__ __forceinline__ unsigned short f2bf(float f) {
  unsigned int u = __float_as_uint(f);
  u += 0x7FFFu + ((u >> 16) & 1u);        // round-to-nearest-even
  return (unsigned short)(u >> 16);
}

__device__ __forceinline__ v8f zero8() {
  v8f z = {0.f, 0.f, 0.f, 0.f, 0.f, 0.f, 0.f, 0.f};
  return z;
}

// CDNA5 async global->LDS copy (ASYNCcnt-tracked, bypasses VGPRs).
// lds_off = LDS byte offset (generic-pointer low 32 bits), gaddr = 64-bit global address.
__device__ __forceinline__ void async_copy_b128(unsigned int lds_off, unsigned long gaddr) {
  asm volatile("global_load_async_to_lds_b128 %0, %1, off" :: "v"(lds_off), "v"(gaddr) : "memory");
}
__device__ __forceinline__ void wait_async0() {
  asm volatile("s_wait_asynccnt 0x0" ::: "memory");
}

// ---------------- weight transpose + bf16 convert: src[K][N] f32 -> dst[N][K] bf16 ----------------
__global__ __launch_bounds__(256) void k_wT(const float* __restrict__ src,
                                            unsigned short* __restrict__ dst,
                                            int K, int N) {
  size_t i = (size_t)blockIdx.x * 256 + threadIdx.x;
  if (i >= (size_t)K * N) return;
  int n = (int)(i % N), k = (int)(i / N);
  dst[(size_t)n * K + k] = f2bf(src[i]);
}

// ---------------- LayerNorm (+ optional shifted window partition) ----------------
// x: (32,384,64,64) f32 ; out: (131072,384) bf16 tokens
__global__ __launch_bounds__(256) void k_ln(const float* __restrict__ x,
                                            const float* __restrict__ g,
                                            const float* __restrict__ bb,
                                            unsigned short* __restrict__ out,
                                            int shifted) {
  const int lane = threadIdx.x & 31;
  const int wave = threadIdx.x >> 5;
  const int token = blockIdx.x * 8 + wave;       // 0..131071
  int b, h, w;
  if (shifted) {
    int win = token >> 6, pos = token & 63;
    b = win >> 6;
    int wloc = win & 63;
    int ah = ((wloc >> 3) << 3) + (pos >> 3);    // shifted-frame coords
    int aw = ((wloc & 7) << 3) + (pos & 7);
    h = (ah + 4) & 63;                           // roll(-ss): source pixel
    w = (aw + 4) & 63;
  } else {
    b = token >> 12; h = (token >> 6) & 63; w = token & 63;
  }
  const float* xp = x + (size_t)b * (384 * 4096) + (size_t)h * 64 + w;
  float v[12];
  float s = 0.f;
  #pragma unroll
  for (int i = 0; i < 12; ++i) { v[i] = xp[(size_t)(lane + i * 32) * 4096]; s += v[i]; }
  #pragma unroll
  for (int o = 16; o >= 1; o >>= 1) s += __shfl_xor(s, o, 32);
  float mu = s * (1.f / 384.f);
  float var = 0.f;
  #pragma unroll
  for (int i = 0; i < 12; ++i) { float d = v[i] - mu; var += d * d; }
  #pragma unroll
  for (int o = 16; o >= 1; o >>= 1) var += __shfl_xor(var, o, 32);
  float rs = rsqrtf(var * (1.f / 384.f) + 1e-5f);
  unsigned short* op = out + (size_t)token * 384;
  #pragma unroll
  for (int i = 0; i < 12; ++i) {
    int c = lane + i * 32;
    op[c] = f2bf((v[i] - mu) * rs * g[c] + bb[c]);
  }
}

// ---------------- generic bf16 WMMA GEMM: D = A(MxK) * Bt(NxK)^T + bias ----------------
// mode 0: store bf16 ; mode 1: store f32 ; mode 2: GELU then bf16
// Double-buffered LDS filled with CDNA5 async global->LDS copies (ASYNCcnt).
#define GBM 128
#define GBN 128
#define GBK 32
#define GLD 48   // padded LDS row stride (ushort); 96B = 16B multiple

__global__ __launch_bounds__(256) void k_gemm(const unsigned short* __restrict__ A,
                                              const unsigned short* __restrict__ Bt,
                                              const float* __restrict__ bias,
                                              unsigned short* __restrict__ outBf,
                                              float* __restrict__ outF,
                                              int M, int N, int K, int mode) {
  __shared__ unsigned short As[2][GBM * GLD];
  __shared__ unsigned short Bs[2][GBN * GLD];
  const int lane = threadIdx.x & 31;
  const int wave = threadIdx.x >> 5;       // 8 waves
  const int wm = wave >> 1, wn = wave & 1; // 4x2 wave grid -> 32x64 per wave
  const int lh = lane >> 4, ll = lane & 15;
  const int bm = blockIdx.x, bn = blockIdx.y;

  v8f acc[2][4];
  #pragma unroll
  for (int i = 0; i < 2; ++i)
    #pragma unroll
    for (int j = 0; j < 4; ++j) acc[i][j] = zero8();

  const size_t aRow0 = (size_t)bm * GBM;
  const size_t bRow0 = (size_t)bn * GBN;

  // per-thread copy slots: 512 16B chunks for A, 512 for B, over 256 threads
  const int id0 = (int)threadIdx.x;
  const int r0 = id0 >> 2, c0 = (id0 & 3) << 3;
  const int id1 = id0 + 256;
  const int r1 = id1 >> 2, c1 = (id1 & 3) << 3;

  auto issue_tile = [&](int buf, int kt) {
    async_copy_b128((unsigned int)(uintptr_t)&As[buf][r0 * GLD + c0],
                    (unsigned long)(uintptr_t)&A[(aRow0 + r0) * K + kt + c0]);
    async_copy_b128((unsigned int)(uintptr_t)&Bs[buf][r0 * GLD + c0],
                    (unsigned long)(uintptr_t)&Bt[(bRow0 + r0) * K + kt + c0]);
    async_copy_b128((unsigned int)(uintptr_t)&As[buf][r1 * GLD + c1],
                    (unsigned long)(uintptr_t)&A[(aRow0 + r1) * K + kt + c1]);
    async_copy_b128((unsigned int)(uintptr_t)&Bs[buf][r1 * GLD + c1],
                    (unsigned long)(uintptr_t)&Bt[(bRow0 + r1) * K + kt + c1]);
  };

  const int nk = K / GBK;
  issue_tile(0, 0);
  wait_async0();
  __syncthreads();

  for (int ki = 0; ki < nk; ++ki) {
    const int cur = ki & 1;
    if (ki + 1 < nk) issue_tile(cur ^ 1, (ki + 1) * GBK);   // prefetch next tile (async)

    FragB16 fa[2], fb[4];
    #pragma unroll
    for (int mi = 0; mi < 2; ++mi) {                 // A frag: two 16B chunks
      const unsigned short* p = &As[cur][(wm * 32 + mi * 16 + ll) * GLD + lh * 8];
      fa[mi].q[0] = *(const uint4*)p;
      fa[mi].q[1] = *(const uint4*)(p + 16);
    }
    #pragma unroll
    for (int ni = 0; ni < 4; ++ni) {                 // B frag: two 16B chunks
      const unsigned short* p = &Bs[cur][(wn * 64 + ni * 16 + ll) * GLD + lh * 8];
      fb[ni].q[0] = *(const uint4*)p;
      fb[ni].q[1] = *(const uint4*)(p + 16);
    }
    #pragma unroll
    for (int mi = 0; mi < 2; ++mi)
      #pragma unroll
      for (int ni = 0; ni < 4; ++ni)
        acc[mi][ni] = __builtin_amdgcn_wmma_f32_16x16x32_bf16(
            false, fa[mi].v, false, fb[ni].v, (short)0, acc[mi][ni], false, false);

    wait_async0();      // next buffer landed (overlapped with the WMMAs above)
    __syncthreads();    // all waves done reading cur (frags consumed by wmma) + next visible
  }

  #pragma unroll
  for (int mi = 0; mi < 2; ++mi)
    #pragma unroll
    for (int ni = 0; ni < 4; ++ni) {
      int gn = bn * GBN + wn * 64 + ni * 16 + ll;
      float bv = bias[gn];
      #pragma unroll
      for (int r = 0; r < 8; ++r) {
        int gm = bm * GBM + wm * 32 + mi * 16 + r + lh * 8;   // C-layout: M = r + 8*(lane>>4)
        float v = acc[mi][ni][r] + bv;
        if (mode == 2) v = 0.5f * v * (1.0f + erff(v * 0.70710678118f));
        size_t o = (size_t)gm * N + gn;
        if (mode == 1) outF[o] = v;
        else outBf[o] = f2bf(v);
      }
    }
}

// ---------------- per-window WMMA attention (one wave per head) ----------------
__device__ __forceinline__ int regionOf(int c) { return c < 56 ? 0 : (c < 60 ? 1 : 2); }

__global__ __launch_bounds__(64) void k_attn(const unsigned short* __restrict__ qkv, // (131072,1152) bf16
                                             const float* __restrict__ rpb,          // (225,12) f32
                                             unsigned short* __restrict__ outA) {    // (131072,384) bf16
  const int VST = 72;   // vT row stride (144B, 16B multiple)
  const int PST = 80;   // P row stride (160B)
  __shared__ unsigned short vt_s[2][32 * 72];
  __shared__ unsigned short ps_s[2][64 * 80];
  const int lane = threadIdx.x & 31;
  const int wave = threadIdx.x >> 5;
  const int head = blockIdx.y * 2 + wave;    // grid.y=6 * 2 waves = 12 heads
  const int win  = blockIdx.x;               // 0..2047
  unsigned short* vt = vt_s[wave];
  unsigned short* ps = ps_s[wave];
  const int lh = lane >> 4, ll = lane & 15;
  const size_t row0 = (size_t)win * 64;
  const int qoff = head * 32, koff = 384 + head * 32, voff = 768 + head * 32;

  // stage V transposed: vt[d][token]
  #pragma unroll
  for (int it = 0; it < 8; ++it) {
    int idx = lane + it * 32;
    int tok = idx >> 2, c8 = (idx & 3) << 3;
    uint4 d = *(const uint4*)&qkv[(row0 + tok) * 1152 + voff + c8];
    const unsigned short* e = (const unsigned short*)&d;
    #pragma unroll
    for (int j = 0; j < 8; ++j) vt[(c8 + j) * VST + tok] = e[j];
  }
  __syncthreads();

  // S = q * k^T (K = hd = 32, one WMMA per 16x16 tile), frags straight from global
  v8f s[4][4];
  #pragma unroll
  for (int i = 0; i < 4; ++i)
    #pragma unroll
    for (int j = 0; j < 4; ++j) s[i][j] = zero8();

  FragB16 fq[4], fk[4];
  #pragma unroll
  for (int mi = 0; mi < 4; ++mi) {
    const unsigned short* p = &qkv[(row0 + mi * 16 + ll) * 1152 + qoff + lh * 8];
    fq[mi].q[0] = *(const uint4*)p;
    fq[mi].q[1] = *(const uint4*)(p + 16);
  }
  #pragma unroll
  for (int ni = 0; ni < 4; ++ni) {
    const unsigned short* p = &qkv[(row0 + ni * 16 + ll) * 1152 + koff + lh * 8];
    fk[ni].q[0] = *(const uint4*)p;
    fk[ni].q[1] = *(const uint4*)(p + 16);
  }
  #pragma unroll
  for (int mi = 0; mi < 4; ++mi)
    #pragma unroll
    for (int ni = 0; ni < 4; ++ni)
      s[mi][ni] = __builtin_amdgcn_wmma_f32_16x16x32_bf16(
          false, fq[mi].v, false, fk[ni].v, (short)0, s[mi][ni], false, false);

  // scale + relative-position bias + shift mask, then softmax on accumulators
  const float scale = 0.17677669529663687f;      // 1/sqrt(32)
  const int wloc = win & 63, wh = wloc >> 3, ww = wloc & 7;
  #pragma unroll
  for (int mi = 0; mi < 4; ++mi) {
    #pragma unroll
    for (int r = 0; r < 8; ++r) {
      int ntok = mi * 16 + r + lh * 8;           // query token (C-layout row)
      int niy = ntok >> 3, njx = ntok & 7;
      int idn = regionOf(wh * 8 + niy) * 3 + regionOf(ww * 8 + njx);
      #pragma unroll
      for (int ni = 0; ni < 4; ++ni) {
        int mtok = ni * 16 + ll;                 // key token (C-layout col)
        int miy = mtok >> 3, mjx = mtok & 7;
        int ridx = (niy - miy + 7) * 15 + (njx - mjx + 7);
        int idm = regionOf(wh * 8 + miy) * 3 + regionOf(ww * 8 + mjx);
        float add = rpb[ridx * 12 + head] + ((idn != idm) ? -100.f : 0.f);
        s[mi][ni][r] = s[mi][ni][r] * scale + add;
      }
      // row softmax: row spans 16 lanes of one half-wave + 4 ni chunks
      float mx = -3.0e38f;
      #pragma unroll
      for (int ni = 0; ni < 4; ++ni) mx = fmaxf(mx, s[mi][ni][r]);
      #pragma unroll
      for (int o = 1; o < 16; o <<= 1) mx = fmaxf(mx, __shfl_xor(mx, o, 32));
      float sum = 0.f;
      #pragma unroll
      for (int ni = 0; ni < 4; ++ni) {
        float e = __expf(s[mi][ni][r] - mx);
        s[mi][ni][r] = e;
        sum += e;
      }
      #pragma unroll
      for (int o = 1; o < 16; o <<= 1) sum += __shfl_xor(sum, o, 32);
      float inv = 1.f / sum;
      int row = mi * 16 + r + lh * 8;
      #pragma unroll
      for (int ni = 0; ni < 4; ++ni)
        ps[row * PST + ni * 16 + ll] = f2bf(s[mi][ni][r] * inv);
    }
  }
  __syncthreads();

  // O = P(64x64) * V(64x32): K = 64 -> two chained WMMAs per tile
  v8f o[4][2];
  #pragma unroll
  for (int i = 0; i < 4; ++i)
    #pragma unroll
    for (int j = 0; j < 2; ++j) o[i][j] = zero8();

  #pragma unroll
  for (int kt = 0; kt < 2; ++kt) {
    FragB16 fp[4], fv[2];
    #pragma unroll
    for (int mi = 0; mi < 4; ++mi) {
      const unsigned short* p = &ps[(mi * 16 + ll) * PST + kt * 32 + lh * 8];
      fp[mi].q[0] = *(const uint4*)p;
      fp[mi].q[1] = *(const uint4*)(p + 16);
    }
    #pragma unroll
    for (int n2 = 0; n2 < 2; ++n2) {
      const unsigned short* p = &vt[(n2 * 16 + ll) * VST + kt * 32 + lh * 8];
      fv[n2].q[0] = *(const uint4*)p;
      fv[n2].q[1] = *(const uint4*)(p + 16);
    }
    #pragma unroll
    for (int mi = 0; mi < 4; ++mi)
      #pragma unroll
      for (int n2 = 0; n2 < 2; ++n2)
        o[mi][n2] = __builtin_amdgcn_wmma_f32_16x16x32_bf16(
            false, fp[mi].v, false, fv[n2].v, (short)0, o[mi][n2], false, false);
  }

  #pragma unroll
  for (int mi = 0; mi < 4; ++mi)
    #pragma unroll
    for (int n2 = 0; n2 < 2; ++n2) {
      int gc = head * 32 + n2 * 16 + ll;
      #pragma unroll
      for (int r = 0; r < 8; ++r) {
        size_t gm = row0 + mi * 16 + r + lh * 8;
        outA[gm * 384 + gc] = f2bf(o[mi][n2][r]);
      }
    }
}

// ---------------- residual add with window-reverse + un-shift scatter ----------------
__global__ __launch_bounds__(256) void k_res1(const float* __restrict__ x,
                                              const float* __restrict__ p,
                                              float* __restrict__ y) {
  size_t i = (size_t)blockIdx.x * 256 + threadIdx.x;  // over 32*384*64*64
  int w = (int)(i & 63), h = (int)((i >> 6) & 63);
  int c = (int)((i >> 12) % 384);
  int b = (int)(i / ((size_t)384 * 4096));
  int ah = (h + 60) & 63, aw = (w + 60) & 63;         // inverse of roll(+ss)
  size_t tok = (size_t)(b * 64 + (ah >> 3) * 8 + (aw >> 3)) * 64 + (size_t)((ah & 7) * 8 + (aw & 7));
  y[i] = x[i] + p[tok * 384 + c];
}

// ---------------- final residual (natural token order) ----------------
__global__ __launch_bounds__(256) void k_final(const float* __restrict__ y,
                                               const float* __restrict__ m,
                                               float* __restrict__ out) {
  size_t i = (size_t)blockIdx.x * 256 + threadIdx.x;
  int w = (int)(i & 63), h = (int)((i >> 6) & 63);
  int c = (int)((i >> 12) % 384);
  int b = (int)(i / ((size_t)384 * 4096));
  size_t tok = (size_t)b * 4096 + (size_t)h * 64 + w;
  out[i] = y[i] + m[tok * 384 + c];
}

// ---------------- host orchestration ----------------
extern "C" void kernel_launch(void* const* d_in, const int* in_sizes, int n_in,
                              void* d_out, int out_size, void* d_ws, size_t ws_size,
                              hipStream_t stream) {
  (void)in_sizes; (void)n_in; (void)out_size; (void)ws_size;
  const float* x      = (const float*)d_in[0];
  const float* qkv_w  = (const float*)d_in[1];
  const float* qkv_b  = (const float*)d_in[2];
  const float* proj_w = (const float*)d_in[3];
  const float* proj_b = (const float*)d_in[4];
  const float* rpb    = (const float*)d_in[5];
  const float* n1g    = (const float*)d_in[6];
  const float* n1b    = (const float*)d_in[7];
  const float* n2g    = (const float*)d_in[8];
  const float* n2b    = (const float*)d_in[9];
  const float* w1     = (const float*)d_in[10];
  const float* b1     = (const float*)d_in[11];
  const float* w2     = (const float*)d_in[12];
  const float* b2     = (const float*)d_in[13];

  char* base = (char*)d_ws;
  size_t off = 0;
  auto take = [&](size_t bytes) -> char* {
    char* p = base + off;
    off = (off + bytes + 255) & ~(size_t)255;
    return p;
  };
  unsigned short* wqkvT  = (unsigned short*)take(1152ull * 384 * 2);
  unsigned short* wprojT = (unsigned short*)take(384ull * 384 * 2);
  unsigned short* w1T    = (unsigned short*)take(1536ull * 384 * 2);
  unsigned short* w2T    = (unsigned short*)take(384ull * 1536 * 2);
  unsigned short* A0     = (unsigned short*)take(131072ull * 384 * 2);   // LN1 out, reused as LN2 out
  unsigned short* QKV    = (unsigned short*)take(131072ull * 1536 * 2);  // QKV out, reused as MLP hidden
  unsigned short* attnO  = (unsigned short*)take(131072ull * 384 * 2);
  float*          f32buf = (float*)take(131072ull * 384 * 4);            // proj out, reused as MLP out
  float*          y      = (float*)take(131072ull * 384 * 4);            // residual-1 result

  // weight prep (bf16, transposed to [N][K])
  k_wT<<<(1152 * 384 + 255) / 256, 256, 0, stream>>>(qkv_w, wqkvT, 384, 1152);
  k_wT<<<(384 * 384 + 255) / 256, 256, 0, stream>>>(proj_w, wprojT, 384, 384);
  k_wT<<<(384 * 1536 + 255) / 256, 256, 0, stream>>>(w1, w1T, 384, 1536);
  k_wT<<<(1536 * 384 + 255) / 256, 256, 0, stream>>>(w2, w2T, 1536, 384);

  // LN1 + shifted window partition
  k_ln<<<16384, 256, 0, stream>>>(x, n1g, n1b, A0, 1);
  // QKV projection: (131072 x 384) @ (384 x 1152)
  k_gemm<<<dim3(1024, 9), 256, 0, stream>>>(A0, wqkvT, qkv_b, QKV, nullptr, 131072, 1152, 384, 0);
  // windowed multi-head attention (WMMA S and O)
  k_attn<<<dim3(2048, 6), 64, 0, stream>>>(QKV, rpb, attnO);
  // output projection (f32 out)
  k_gemm<<<dim3(1024, 3), 256, 0, stream>>>(attnO, wprojT, proj_b, nullptr, f32buf, 131072, 384, 384, 1);
  // window reverse + un-shift + residual
  k_res1<<<196608, 256, 0, stream>>>(x, f32buf, y);
  // LN2 (natural token order)
  k_ln<<<16384, 256, 0, stream>>>(y, n2g, n2b, A0, 0);
  // MLP: fc1 + exact GELU (bf16 out), fc2 (f32 out)
  k_gemm<<<dim3(1024, 12), 256, 0, stream>>>(A0, w1T, b1, QKV, nullptr, 131072, 1536, 384, 2);
  k_gemm<<<dim3(1024, 3), 256, 0, stream>>>(QKV, w2T, b2, nullptr, f32buf, 131072, 384, 1536, 1);
  // final residual
  k_final<<<196608, 256, 0, stream>>>(y, f32buf, (float*)d_out);
}
